// GlobalLightningAttention_34565896798846
// MI455X (gfx1250) — matrix operand order repaired and checked by reading, not compiled, not run on hardware
//
#include <hip/hip_runtime.h>
#include <hip/hip_bf16.h>

// ---------------------------------------------------------------------------
// Types for CDNA5 (gfx1250) wave32 WMMA / TDM
// ---------------------------------------------------------------------------
typedef __bf16 bf16_t;
typedef bf16_t bf16x16 __attribute__((ext_vector_type(16)));
typedef float  f32x8   __attribute__((ext_vector_type(8)));
typedef unsigned int u32x4 __attribute__((ext_vector_type(4)));
typedef int          i32x4 __attribute__((ext_vector_type(4)));
typedef int          i32x8 __attribute__((ext_vector_type(8)));

union FragU { bf16x16 v; uint4 q[2]; };

// A-matrix fragment, 16x32 bf16 (MxK):
//   lane<16 : row=lane,    K = [0..15]
//   lane>=16: row=lane-16, K = [16..31]
__device__ __forceinline__ bf16x16 load_frag_a(const bf16_t* __restrict__ base,
                                               int ld, int lane) {
  const int r = lane & 15;
  const int h = lane >> 4;
  const bf16_t* p = base + (size_t)r * ld + h * 16;
  FragU u;
  u.q[0] = *reinterpret_cast<const uint4*>(p);
  u.q[1] = *reinterpret_cast<const uint4*>(p + 8);
  return u.v;
}

// B-matrix fragment, 32x16 bf16 (KxN), loaded from B^T stored row-major [N,K]:
//   lane<16 : col=lane,    K = [0..15] ; lane>=16: col=lane-16, K = [16..31]
// Works for both global and LDS source pointers.
__device__ __forceinline__ bf16x16 load_frag_bT(const bf16_t* __restrict__ base,
                                                int ld, int lane) {
  const int c = lane & 15;
  const int h = lane >> 4;
  const bf16_t* p = base + (size_t)c * ld + h * 16;
  FragU u;
  u.q[0] = reinterpret_cast<const uint4*>(p)[0];
  u.q[1] = reinterpret_cast<const uint4*>(p)[1];
  return u.v;
}

__device__ __forceinline__ f32x8 wmma_bf16(bf16x16 a, bf16x16 b, f32x8 c) {
  // (neg_a, A, neg_b, B, c_mod, C, reuse_a, reuse_b)
  return __builtin_amdgcn_wmma_f32_16x16x32_bf16(false, a, false, b,
                                                 (short)0, c, false, false);
}

// ---------------------------------------------------------------------------
// TDM: DMA a [64 rows x 128 cols] bf16 tile (row stride = tensor_k elements)
// from global memory into LDS at byte offset lds_off (row-major, ld=128).
// Descriptor layout per CDNA5 ISA.
// ---------------------------------------------------------------------------
__device__ __forceinline__ void tdm_load_b_tile(const bf16_t* gsrc,
                                                unsigned lds_off,
                                                int tensor_k, int tensor_n) {
  const unsigned long long ga = (unsigned long long)(uintptr_t)gsrc;
  u32x4 g0;
  g0[0] = 1u;                                   // count=1, user descriptor
  g0[1] = lds_off;                              // lds_addr (bytes)
  g0[2] = (unsigned)ga;                         // global_addr[31:0]
  g0[3] = (unsigned)(ga >> 32) | 0x80000000u;   // addr[56:32] | type=2<<30
  i32x8 g1;
  g1[0] = 0x00010000;                           // data_size=1 (2B); mask=0
  g1[1] = (tensor_k & 0xffff) << 16;            // tensor_dim0[15:0]
  g1[2] = ((tensor_k >> 16) & 0xffff) |         // tensor_dim0[31:16]
          ((tensor_n & 0xffff) << 16);          // tensor_dim1[15:0]
  g1[3] = ((tensor_n >> 16) & 0xffff) |         // tensor_dim1[31:16]
          (128 << 16);                          // tile_dim0 = 128
  g1[4] = 64;                                   // tile_dim1 = 64, tile_dim2=0
  g1[5] = tensor_k;                             // tensor_dim0_stride[31:0]
  g1[6] = 0;                                    // stride0[47:32], stride1 lo
  g1[7] = 0;
  i32x4 gz = {0, 0, 0, 0};
#if defined(__clang_major__) && __clang_major__ >= 23
  i32x8 gz8 = {0, 0, 0, 0, 0, 0, 0, 0};
  __builtin_amdgcn_tensor_load_to_lds(g0, g1, gz, gz, gz8, 0);
#else
  __builtin_amdgcn_tensor_load_to_lds(g0, g1, gz, gz, 0);
#endif
}

// ---------------------------------------------------------------------------
// fp32 -> bf16 convert
// ---------------------------------------------------------------------------
__global__ void cvt_f32_to_bf16(const float* __restrict__ src,
                                bf16_t* __restrict__ dst, int n) {
  int i = blockIdx.x * blockDim.x + threadIdx.x;
  if (i < n) dst[i] = (bf16_t)src[i];
}

// ---------------------------------------------------------------------------
// C[M,N] = A[M,K] (bf16) x B^T[N,K] (bf16)   (i.e. y = x @ W^T)
// Block: 256 thr = 8 waves, each wave a 32x64 tile; block covers 256x64.
// B (shared by all 8 waves) is staged into LDS by the Tensor Data Mover in
// double-buffered 64x128 slabs; waves consume it via ds_load_b128.
// ---------------------------------------------------------------------------
template <bool STORE_F32>
__global__ void __launch_bounds__(256) gemm_bf16_wT(
    const bf16_t* __restrict__ A, const bf16_t* __restrict__ BT,
    void* __restrict__ Cout, int M, int N, int K) {
  __shared__ bf16_t Bs[2][64 * 128];  // 2 x 16 KB

  const int lane = threadIdx.x & 31;
  const int wave = threadIdx.x >> 5;
  const int m0 = blockIdx.x * 256 + wave * 32;
  const int n0 = blockIdx.y * 64;

  f32x8 acc[2][4];
#pragma unroll
  for (int i = 0; i < 2; ++i)
#pragma unroll
    for (int j = 0; j < 4; ++j) acc[i][j] = {};

  const bf16_t* A0 = A + (size_t)m0 * K;
  const bf16_t* A1 = A + (size_t)(m0 + 16) * K;
  const bf16_t* B0 = BT + (size_t)n0 * K;

  // Preload first B slab via TDM (wave 0 only; TDM ignores EXEC).
  if (wave == 0)
    tdm_load_b_tile(B0, (unsigned)(uintptr_t)&Bs[0][0], K, N);

  const int nslab = K >> 7;  // 128-wide K slabs
  for (int slab = 0; slab < nslab; ++slab) {
    const int p = slab & 1;
    const int k0 = slab << 7;
    if (wave == 0) __builtin_amdgcn_s_wait_tensorcnt(0);
    __syncthreads();  // slab p ready; previous readers of p^1 are done
    if (wave == 0 && slab + 1 < nslab)
      tdm_load_b_tile(B0 + k0 + 128, (unsigned)(uintptr_t)&Bs[p ^ 1][0], K, N);

    const bf16_t* Bb = &Bs[p][0];
#pragma unroll
    for (int kk = 0; kk < 128; kk += 32) {
      __builtin_prefetch(A0 + k0 + kk + 256, 0, 0);  // next A tiles
      bf16x16 a0 = load_frag_a(A0 + k0 + kk, K, lane);
      bf16x16 a1 = load_frag_a(A1 + k0 + kk, K, lane);
      bf16x16 b0 = load_frag_bT(Bb + kk, 128, lane);
      bf16x16 b1 = load_frag_bT(Bb + 16 * 128 + kk, 128, lane);
      bf16x16 b2 = load_frag_bT(Bb + 32 * 128 + kk, 128, lane);
      bf16x16 b3 = load_frag_bT(Bb + 48 * 128 + kk, 128, lane);
      acc[0][0] = wmma_bf16(a0, b0, acc[0][0]);
      acc[0][1] = wmma_bf16(a0, b1, acc[0][1]);
      acc[0][2] = wmma_bf16(a0, b2, acc[0][2]);
      acc[0][3] = wmma_bf16(a0, b3, acc[0][3]);
      acc[1][0] = wmma_bf16(a1, b0, acc[1][0]);
      acc[1][1] = wmma_bf16(a1, b1, acc[1][1]);
      acc[1][2] = wmma_bf16(a1, b2, acc[1][2]);
      acc[1][3] = wmma_bf16(a1, b3, acc[1][3]);
    }
  }

  // C/D layout: VGPR v, lane -> (M = v + 8*(lane>>4), N = lane&15)
  const int half = lane >> 4;
  const int cN = lane & 15;
#pragma unroll
  for (int i = 0; i < 2; ++i)
#pragma unroll
    for (int j = 0; j < 4; ++j)
#pragma unroll
      for (int v = 0; v < 8; ++v) {
        const int row = m0 + 16 * i + v + 8 * half;
        const int col = n0 + 16 * j + cN;
        if constexpr (STORE_F32)
          reinterpret_cast<float*>(Cout)[(size_t)row * N + col] = acc[i][j][v];
        else
          reinterpret_cast<bf16_t*>(Cout)[(size_t)row * N + col] =
              (bf16_t)acc[i][j][v];
      }
}

// ---------------------------------------------------------------------------
// Fused LayerNorm + RoPE + SiLU + head split/transpose.
// One block per token (b,l); 256 threads handle 1024 features.
//   q_s [BH, L, 64]   (A operand of out-GEMM)
//   k_s [BH, 64, L]   (A operand of kv-GEMM, i.e. k^T per head)
//   v_t [BH, 64, L]   (B^T operand of kv-GEMM)
// ---------------------------------------------------------------------------
__global__ void __launch_bounds__(256) ln_rope_silu_kernel(
    const bf16_t* __restrict__ Qb, const bf16_t* __restrict__ Kb,
    const bf16_t* __restrict__ Vb, const float* __restrict__ gq,
    const float* __restrict__ bq, const float* __restrict__ gk,
    const float* __restrict__ bk, bf16_t* __restrict__ qs,
    bf16_t* __restrict__ ks, bf16_t* __restrict__ vt, int L) {
  __shared__ float row[1024];
  __shared__ float red[512];

  const int token = blockIdx.x;
  const int b = token / L;
  const int l = token - b * L;
  const int tid = threadIdx.x;
  const float inv1024 = 1.0f / 1024.0f;

  for (int p = 0; p < 2; ++p) {
    const bf16_t* src = p ? Kb : Qb;
    const float* g = p ? gk : gq;
    const float* bb = p ? bk : bq;

    float vals[4];
    float s = 0.f, sq = 0.f;
#pragma unroll
    for (int j = 0; j < 4; ++j) {
      const int f = tid + 256 * j;
      const float x = (float)src[(size_t)token * 1024 + f];
      vals[j] = x;
      s += x;
      sq += x * x;
    }
    red[tid] = s;
    red[256 + tid] = sq;
    __syncthreads();
    for (int st = 128; st > 0; st >>= 1) {
      if (tid < st) {
        red[tid] += red[tid + st];
        red[256 + tid] += red[256 + tid + st];
      }
      __syncthreads();
    }
    const float mean = red[0] * inv1024;
    const float var = red[256] * inv1024 - mean * mean;
    const float rstd = rsqrtf(var + 1e-5f);

#pragma unroll
    for (int j = 0; j < 4; ++j) {
      const int f = tid + 256 * j;
      row[f] = (vals[j] - mean) * rstd * g[f] + bb[f];
    }
    __syncthreads();

#pragma unroll
    for (int j = 0; j < 4; ++j) {
      const int f = tid + 256 * j;
      const int h = f >> 6;
      const int d = f & 63;
      const int jj = d & 31;
      const float freq = __powf(10000.0f, -(float)(2 * jj) * (1.0f / 64.0f));
      const float ang = (float)l * freq;
      float sn, cs;
      __sincosf(ang, &sn, &cs);
      const float x = row[f];
      const float xp = (d < 32) ? row[f + 32] : row[f - 32];
      const float r = (d < 32) ? (x * cs - xp * sn) : (x * cs + xp * sn);
      const float sl = r / (1.0f + __expf(-r));
      const size_t bh = (size_t)(b * 16 + h);
      if (p == 0)
        qs[(bh * L + l) * 64 + d] = (bf16_t)sl;
      else
        ks[(bh * 64 + d) * (size_t)L + l] = (bf16_t)sl;
    }
    __syncthreads();
  }

  // V: no LN / rope, just split heads + transpose to [BH, 64, L]
#pragma unroll
  for (int j = 0; j < 4; ++j) {
    const int f = tid + 256 * j;
    const int h = f >> 6;
    const int d = f & 63;
    const float x = (float)Vb[(size_t)token * 1024 + f];
    const size_t bh = (size_t)(b * 16 + h);
    vt[(bh * 64 + d) * (size_t)L + l] = (bf16_t)x;
  }
}

// ---------------------------------------------------------------------------
// kv^T[e,d] = sum_l k[l,d] * v[l,e]  per (b,h).
// 64 blocks (one per b,h), 512 thr = 16 waves; wave -> one 16x16 (d,e) tile.
// ---------------------------------------------------------------------------
__global__ void __launch_bounds__(512) kv_kernel(const bf16_t* __restrict__ ks,
                                                 const bf16_t* __restrict__ vt,
                                                 bf16_t* __restrict__ kvt,
                                                 int L) {
  const int bh = blockIdx.x;
  const int lane = threadIdx.x & 31;
  const int wave = threadIdx.x >> 5;
  const int dt = (wave >> 2) * 16;
  const int et = (wave & 3) * 16;

  const bf16_t* Ak = ks + (size_t)bh * 64 * L + (size_t)dt * L;
  const bf16_t* Bv = vt + (size_t)bh * 64 * L + (size_t)et * L;

  f32x8 acc = {};
  for (int k0 = 0; k0 < L; k0 += 32) {
    __builtin_prefetch(Ak + k0 + 128, 0, 0);
    bf16x16 a = load_frag_a(Ak + k0, L, lane);
    bf16x16 b = load_frag_bT(Bv + k0, L, lane);
    acc = wmma_bf16(a, b, acc);
  }

  const int half = lane >> 4;
  const int e = et + (lane & 15);
  bf16_t* out = kvt + (size_t)bh * 4096;
#pragma unroll
  for (int v = 0; v < 8; ++v) {
    const int d = dt + v + 8 * half;
    out[(size_t)e * 64 + d] = (bf16_t)acc[v];
  }
}

// ---------------------------------------------------------------------------
// att[b,l,h*64+e] = (1/8) * sum_d q_s[bh,l,d] * kv[d,e]
// ---------------------------------------------------------------------------
__global__ void __launch_bounds__(256) attn_out_kernel(
    const bf16_t* __restrict__ qs, const bf16_t* __restrict__ kvt,
    bf16_t* __restrict__ att, int L) {
  const int bh = blockIdx.y;
  const int b = bh >> 4;
  const int h = bh & 15;
  const int lane = threadIdx.x & 31;
  const int wave = threadIdx.x >> 5;
  const int m0 = blockIdx.x * 256 + wave * 32;

  const bf16_t* A = qs + (size_t)bh * L * 64;
  const bf16_t* BT = kvt + (size_t)bh * 4096;  // [e,d] row-major 64x64

  f32x8 acc[2][4];
#pragma unroll
  for (int i = 0; i < 2; ++i)
#pragma unroll
    for (int j = 0; j < 4; ++j) acc[i][j] = {};

#pragma unroll
  for (int k0 = 0; k0 < 64; k0 += 32) {
    bf16x16 a0 = load_frag_a(A + (size_t)m0 * 64 + k0, 64, lane);
    bf16x16 a1 = load_frag_a(A + (size_t)(m0 + 16) * 64 + k0, 64, lane);
#pragma unroll
    for (int j = 0; j < 4; ++j) {
      bf16x16 bj = load_frag_bT(BT + (size_t)(j * 16) * 64 + k0, 64, lane);
      acc[0][j] = wmma_bf16(a0, bj, acc[0][j]);
      acc[1][j] = wmma_bf16(a1, bj, acc[1][j]);
    }
  }

  const int half = lane >> 4;
  const int cN = lane & 15;
#pragma unroll
  for (int i = 0; i < 2; ++i)
#pragma unroll
    for (int j = 0; j < 4; ++j)
#pragma unroll
      for (int v = 0; v < 8; ++v) {
        const int l = m0 + 16 * i + v + 8 * half;
        const int e = 16 * j + cN;
        att[((size_t)b * L + l) * 1024 + h * 64 + e] =
            (bf16_t)(0.125f * acc[i][j][v]);
      }
}

// ---------------------------------------------------------------------------
// Host launcher
// ---------------------------------------------------------------------------
extern "C" void kernel_launch(void* const* d_in, const int* in_sizes, int n_in,
                              void* d_out, int out_size, void* d_ws,
                              size_t ws_size, hipStream_t stream) {
  const float* x = (const float*)d_in[0];
  const float* wq = (const float*)d_in[1];
  const float* wk = (const float*)d_in[2];
  const float* wv = (const float*)d_in[3];
  const float* wo = (const float*)d_in[4];
  const float* gq = (const float*)d_in[5];
  const float* bq = (const float*)d_in[6];
  const float* gk = (const float*)d_in[7];
  const float* bk = (const float*)d_in[8];

  const int B = 4, L = 4096, D = 1024;
  const int M = B * L;  // 16384 tokens

  char* ws = (char*)d_ws;
  size_t off = 0;
  auto take = [&](size_t bytes) -> char* {
    char* p = ws + off;
    off = (off + bytes + 255) & ~(size_t)255;
    return p;
  };

  bf16_t* xb = (bf16_t*)take((size_t)M * D * sizeof(bf16_t));
  bf16_t* wqb = (bf16_t*)take((size_t)D * D * sizeof(bf16_t));
  bf16_t* wkb = (bf16_t*)take((size_t)D * D * sizeof(bf16_t));
  bf16_t* wvb = (bf16_t*)take((size_t)D * D * sizeof(bf16_t));
  bf16_t* wob = (bf16_t*)take((size_t)D * D * sizeof(bf16_t));
  bf16_t* Qb = (bf16_t*)take((size_t)M * D * sizeof(bf16_t));
  bf16_t* Kb = (bf16_t*)take((size_t)M * D * sizeof(bf16_t));
  bf16_t* Vb = (bf16_t*)take((size_t)M * D * sizeof(bf16_t));
  bf16_t* qs = (bf16_t*)take((size_t)M * D * sizeof(bf16_t));
  bf16_t* ks = (bf16_t*)take((size_t)M * D * sizeof(bf16_t));
  bf16_t* vt = (bf16_t*)take((size_t)M * D * sizeof(bf16_t));
  bf16_t* kvt = (bf16_t*)take((size_t)B * 16 * 64 * 64 * sizeof(bf16_t));
  bf16_t* att = Qb;  // Qb is fully consumed before attn_out writes att

  // 1) fp32 -> bf16 converts
  {
    const int n = M * D;
    cvt_f32_to_bf16<<<dim3((n + 255) / 256), dim3(256), 0, stream>>>(x, xb, n);
    const int nw = D * D;
    dim3 gw((nw + 255) / 256), bw(256);
    cvt_f32_to_bf16<<<gw, bw, 0, stream>>>(wq, wqb, nw);
    cvt_f32_to_bf16<<<gw, bw, 0, stream>>>(wk, wkb, nw);
    cvt_f32_to_bf16<<<gw, bw, 0, stream>>>(wv, wvb, nw);
    cvt_f32_to_bf16<<<gw, bw, 0, stream>>>(wo, wob, nw);
  }

  // 2) Q/K/V projections: [16384,1024] x [1024,1024]^T (TDM-staged B)
  dim3 gg(M / 256, D / 64), bg(256);
  gemm_bf16_wT<false><<<gg, bg, 0, stream>>>(xb, wqb, (void*)Qb, M, D, D);
  gemm_bf16_wT<false><<<gg, bg, 0, stream>>>(xb, wkb, (void*)Kb, M, D, D);
  gemm_bf16_wT<false><<<gg, bg, 0, stream>>>(xb, wvb, (void*)Vb, M, D, D);

  // 3) LN + RoPE + SiLU + head split/transpose
  ln_rope_silu_kernel<<<dim3(M), dim3(256), 0, stream>>>(
      Qb, Kb, Vb, gq, bq, gk, bk, qs, ks, vt, L);

  // 4) kv^T per (b,h)
  kv_kernel<<<dim3(B * 16), dim3(512), 0, stream>>>(ks, vt, kvt, L);

  // 5) out = q @ kv / sqrt(64), merged-head bf16 into att (aliases Qb)
  attn_out_kernel<<<dim3(L / 256, B * 16), dim3(256), 0, stream>>>(qs, kvt, att,
                                                                   L);

  // 6) final projection: att @ wo^T -> fp32 d_out
  gemm_bf16_wT<true><<<gg, bg, 0, stream>>>(att, wob, d_out, M, D, D);
}
